// VisionRetentionChunk_88596585382276
// MI455X (gfx1250) — matrix operand order repaired and checked by reading, compiled, not run on hardware
//
#include <hip/hip_runtime.h>
#include <math.h>

// ---------------------------------------------------------------------------
// CDNA5 (gfx1250) split-bf16 ("3xBF16") matrix path:
//   fp32 operand x = hi(bf16) + lo(bf16);  A*B ~= Ah*Bh + Ah*Bl + Al*Bh
//   -> 3x V_WMMA_F32_16X16X32_BF16 replace 8x V_WMMA_F32_16X16X4_F32 (2.67x)
// Layouts (ISA 7.12.2, wave32):
//   A 16x32 bf16: lanes0-15 hold K {k0..k0+7, k0+16..k0+23}; lanes16-31 +8
//   B 32x16 bf16: lane n&15 = column; K as A: {k0..k0+7, k0+16..k0+23}, lanes16-31 +8
//   C/D 16x16 f32: VGPR r: lanes0-15 (M=r), lanes16-31 (M=r+8), N=lane&15
// ---------------------------------------------------------------------------
typedef __attribute__((ext_vector_type(8)))  float   v8f;
typedef __attribute__((ext_vector_type(8)))  __bf16  v8bf;
typedef __attribute__((ext_vector_type(16))) __bf16  v16bf;

#define E_DIM 256
#define H_DIM 64
#define W_DIM 64

__device__ __forceinline__ v8f wmma_bf16(v16bf a, v16bf b, v8f c) {
  return __builtin_amdgcn_wmma_f32_16x16x32_bf16(
      false, a, false, b, (short)0, c, false, false);
}

// acc += Ah*Bl + Al*Bh + Ah*Bh   (small terms first)
__device__ __forceinline__ v8f mma3(v16bf ah, v16bf al, v16bf bh, v16bf bl,
                                    v8f c) {
  c = wmma_bf16(ah, bl, c);
  c = wmma_bf16(al, bh, c);
  c = wmma_bf16(ah, bh, c);
  return c;
}

__device__ __forceinline__ void split1(float v, __bf16* __restrict__ ph,
                                       __bf16* __restrict__ pl, size_t idx) {
  const __bf16 h = (__bf16)v;
  ph[idx] = h;
  pl[idx] = (__bf16)(v - (float)h);
}

// A fragment: rows = M (row0 + lane&15), reduction dim contiguous in memory.
__device__ __forceinline__ v16bf load_A_bf(const __bf16* __restrict__ p,
                                           int row0, int rowStride, int k0,
                                           int lane) {
  const int r  = row0 + (lane & 15);
  const int b0 = k0 + ((lane >> 4) << 3);  // +0 / +8
  const __bf16* q = p + (size_t)r * rowStride + b0;
  const v8bf x0 = *(const v8bf*)(q);        // K = b0 .. b0+7
  const v8bf x1 = *(const v8bf*)(q + 16);   // K = b0+16 .. b0+23
  v16bf res;
#pragma unroll
  for (int i = 0; i < 8; ++i) { res[i] = x0[i]; res[i + 8] = x1[i]; }
  return res;
}

// B fragment from [N][K] storage (transposed weights / K rows / V^T):
// two 16B runs per lane (K = b0..b0+7, b0+16..b0+23), matching load_A_bf.
__device__ __forceinline__ v16bf load_B_bf(const __bf16* __restrict__ p,
                                           int col0, int rowStride, int k0,
                                           int lane) {
  const int c  = col0 + (lane & 15);
  const int b0 = k0 + ((lane >> 4) << 3);  // +0 / +8
  const __bf16* q = p + (size_t)c * rowStride + b0;
  const v8bf x0 = *(const v8bf*)(q);
  const v8bf x1 = *(const v8bf*)(q + 16);
  v16bf res;
#pragma unroll
  for (int i = 0; i < 8; ++i) { res[i] = x0[i]; res[i + 8] = x1[i]; }
  return res;
}

// ---------------------------------------------------------------------------
// Prep: transpose + hi/lo-split a 256x256 fp32 weight into bf16 [N][K].
// ---------------------------------------------------------------------------
__global__ __launch_bounds__(256) void wprep(const float* __restrict__ W,
                                             __bf16* __restrict__ Th,
                                             __bf16* __restrict__ Tl) {
  const int i = blockIdx.x * 256 + threadIdx.x;  // 65536
  const int k = i >> 8, c = i & 255;
  split1(W[(size_t)k * E_DIM + c], Th, Tl, (size_t)c * E_DIM + k);
}

// Prep: hi/lo-split a [65536 x 256] fp32 tensor (optionally sum of two).
__global__ __launch_bounds__(256) void split_f32(const float* __restrict__ a,
                                                 const float* __restrict__ b,
                                                 __bf16* __restrict__ oh,
                                                 __bf16* __restrict__ ol,
                                                 int hasB) {
  const size_t i = (size_t)blockIdx.x * 256 + threadIdx.x;
  float v = a[i];
  if (hasB) v += b[i];
  split1(v, oh, ol, i);
}

// ---------------------------------------------------------------------------
// Fused GEMM: out[M x 256] = (Ah+Al) @ (Bh+Bl)  + bias, with epilogue
//   MODE 0: q -> theta_shift, split-store     MODE 1: k -> theta*scale, split
//   MODE 2: v -> bias, split-store            MODE 3: out -> bias, fp32 store
// Block 256 thr = 8 waves; block tile 128x64; wave 16x64; K chunks of 32.
// ---------------------------------------------------------------------------
template <int MODE>
__global__ __launch_bounds__(256) void gemm_bf16x3(
    const __bf16* __restrict__ Ah, const __bf16* __restrict__ Al,
    const __bf16* __restrict__ Bh, const __bf16* __restrict__ Bl,
    const float* __restrict__ bias, const float* __restrict__ sinp,
    const float* __restrict__ cosp, __bf16* __restrict__ outh,
    __bf16* __restrict__ outl, float* __restrict__ outf, float scale) {
  const int lane = threadIdx.x & 31;
  const int wv   = threadIdx.x >> 5;
  const int colBase = blockIdx.y * 64;
  const int row0 = blockIdx.x * 128 + wv * 16;

  v8f acc[4] = {};
  for (int k0 = 0; k0 < E_DIM; k0 += 32) {
    const v16bf ah = load_A_bf(Ah, row0, E_DIM, k0, lane);
    const v16bf al = load_A_bf(Al, row0, E_DIM, k0, lane);
#pragma unroll
    for (int t = 0; t < 4; ++t) {
      const v16bf bh = load_B_bf(Bh, colBase + t * 16, E_DIM, k0, lane);
      const v16bf bl = load_B_bf(Bl, colBase + t * 16, E_DIM, k0, lane);
      acc[t] = mma3(ah, al, bh, bl, acc[t]);
    }
  }

#pragma unroll
  for (int t = 0; t < 4; ++t) {
#pragma unroll
    for (int r = 0; r < 8; ++r) {
      const int m = row0 + r + ((lane >> 4) << 3);
      const int c = colBase + t * 16 + (lane & 15);
      float val = acc[t][r] + bias[c];
      if (MODE == 0 || MODE == 1) {
        // theta_shift: val*cos + rot*sin, rot[even]=-x[c+1], rot[odd]=x[c-1]
        const float partner = __shfl_xor(val, 1, 32);
        const float rot = (c & 1) ? partner : -partner;
        const int dd = c & 31;
        const int w  = m & 63;
        const int h  = (m >> 6) & 63;
        const int si = (h * W_DIM + w) * 32 + dd;
        val = val * cosp[si] + rot * sinp[si];
        if (MODE == 1) val *= scale;
      }
      const size_t idx = (size_t)m * E_DIM + c;
      if (MODE == 3) outf[idx] = val;
      else           split1(val, outh, outl, idx);
    }
  }
}

// ---------------------------------------------------------------------------
// Depthwise 5x5 conv (lepe), padding 2; v reconstructed from hi+lo.
// ---------------------------------------------------------------------------
__global__ __launch_bounds__(256) void lepe_conv(
    const __bf16* __restrict__ vh, const __bf16* __restrict__ vl,
    const float* __restrict__ wt, const float* __restrict__ bs,
    float* __restrict__ out) {
  const int idx = blockIdx.x * 256 + threadIdx.x;
  const int c   = idx & 255;
  const int pos = idx >> 8;
  const int w   = pos & 63;
  const int h   = (pos >> 6) & 63;
  const int b   = pos >> 12;
  float acc = bs[c];
  for (int dy = 0; dy < 5; ++dy) {
    const int yy = h + dy - 2;
    if (yy < 0 || yy >= H_DIM) continue;
    for (int dx = 0; dx < 5; ++dx) {
      const int xx = w + dx - 2;
      if (xx < 0 || xx >= W_DIM) continue;
      const size_t g = (((size_t)(b * H_DIM + yy)) * W_DIM + xx) * E_DIM + c;
      acc += ((float)vh[g] + (float)vl[g]) * wt[(dy * 5 + dx) * E_DIM + c];
    }
  }
  out[idx] = acc;
}

// ---------------------------------------------------------------------------
// Batched 64-length attention, split-bf16 everywhere:
//   S = Q K^T + mask ; P = softmax_k(S) ; O = P V
// One workgroup per (b, outer, head); 4 waves own 16 query rows each.
// V^T staged hi/lo in LDS; P split hi/lo into per-wave LDS strips.
// ALONG_W: rows = w (stride 256), split-store v1.  else rows = h (stride
// 16384), fp32 store attn2.
// ---------------------------------------------------------------------------
template <bool ALONG_W>
__global__ __launch_bounds__(128) void attn64(
    const __bf16* __restrict__ qh, const __bf16* __restrict__ ql,
    const __bf16* __restrict__ kh, const __bf16* __restrict__ kl,
    const __bf16* __restrict__ vh, const __bf16* __restrict__ vl,
    const float* __restrict__ maskp, __bf16* __restrict__ o_h,
    __bf16* __restrict__ o_l, float* __restrict__ o_f) {
  const int n     = blockIdx.x & 7;
  const int outer = (blockIdx.x >> 3) & 63;
  const int b     = blockIdx.x >> 9;

  size_t base;
  int rs;
  if (ALONG_W) {
    base = ((size_t)(b * H_DIM + outer)) * W_DIM * E_DIM + n * 32;
    rs   = E_DIM;
  } else {
    base = (((size_t)b * H_DIM) * W_DIM + outer) * E_DIM + n * 32;
    rs   = W_DIM * E_DIM;
  }
  const float* __restrict__ mk = maskp + n * 64 * 64;

  __shared__ __align__(32) __bf16 lvth[32 * 64];  // V^T hi  [d][k]
  __shared__ __align__(32) __bf16 lvtl[32 * 64];  // V^T lo
  __shared__ __align__(32) __bf16 lph[4 * 16 * 64];  // P hi per wave
  __shared__ __align__(32) __bf16 lpl[4 * 16 * 64];  // P lo per wave

  for (int i = threadIdx.x; i < 2048; i += 128) {
    const int krow = i >> 5, d = i & 31;
    const size_t g = base + (size_t)krow * rs + d;
    lvth[d * 64 + krow] = vh[g];
    lvtl[d * 64 + krow] = vl[g];
  }
  __syncthreads();

  const int lane = threadIdx.x & 31;
  const int wv   = threadIdx.x >> 5;
  const int q0   = wv * 16;

  // ---- scores: single K-chunk (kd = 32), 4 col-tiles, 3 wmma each ----
  v8f acc[4] = {};
  {
    const v16bf ah = load_A_bf(qh + base, q0, rs, 0, lane);
    const v16bf al = load_A_bf(ql + base, q0, rs, 0, lane);
#pragma unroll
    for (int t = 0; t < 4; ++t) {
      const v16bf bh = load_B_bf(kh + base, t * 16, rs, 0, lane);
      const v16bf bl = load_B_bf(kl + base, t * 16, rs, 0, lane);
      acc[t] = mma3(ah, al, bh, bl, acc[t]);
    }
  }

  // ---- + mask, row softmax (rows split lower/upper half-wave) ----
#pragma unroll
  for (int t = 0; t < 4; ++t)
#pragma unroll
    for (int r = 0; r < 8; ++r) {
      const int ml = r + ((lane >> 4) << 3);
      const int kc = t * 16 + (lane & 15);
      acc[t][r] += mk[(q0 + ml) * 64 + kc];
    }
#pragma unroll
  for (int r = 0; r < 8; ++r) {
    float m = fmaxf(fmaxf(acc[0][r], acc[1][r]), fmaxf(acc[2][r], acc[3][r]));
    for (int off = 1; off < 16; off <<= 1) m = fmaxf(m, __shfl_xor(m, off, 32));
    float s = 0.f;
#pragma unroll
    for (int t = 0; t < 4; ++t) {
      const float e = __expf(acc[t][r] - m);
      acc[t][r] = e;
      s += e;
    }
    for (int off = 1; off < 16; off <<= 1) s += __shfl_xor(s, off, 32);
    const float inv = 1.0f / s;
#pragma unroll
    for (int t = 0; t < 4; ++t) acc[t][r] *= inv;
  }

  // ---- spill split P strip (per-wave region, same-wave reuse) ----
#pragma unroll
  for (int t = 0; t < 4; ++t)
#pragma unroll
    for (int r = 0; r < 8; ++r) {
      const int ml = r + ((lane >> 4) << 3);
      const int kc = t * 16 + (lane & 15);
      const int li = wv * 1024 + ml * 64 + kc;
      const float pv = acc[t][r];
      const __bf16 hh = (__bf16)pv;
      lph[li] = hh;
      lpl[li] = (__bf16)(pv - (float)hh);
    }

  // ---- O = P V : 2 col-tiles (d), 2 K-chunks (64 keys), 3 wmma each ----
  v8f oacc[2] = {};
#pragma unroll
  for (int c0 = 0; c0 < 64; c0 += 32) {
    const v16bf pah = load_A_bf(lph + wv * 1024, 0, 64, c0, lane);
    const v16bf pal = load_A_bf(lpl + wv * 1024, 0, 64, c0, lane);
#pragma unroll
    for (int t = 0; t < 2; ++t) {
      const v16bf bh = load_B_bf(lvth, t * 16, 64, c0, lane);
      const v16bf bl = load_B_bf(lvtl, t * 16, 64, c0, lane);
      oacc[t] = mma3(pah, pal, bh, bl, oacc[t]);
    }
  }
#pragma unroll
  for (int t = 0; t < 2; ++t)
#pragma unroll
    for (int r = 0; r < 8; ++r) {
      const int qrow = q0 + r + ((lane >> 4) << 3);
      const int d    = t * 16 + (lane & 15);
      const size_t oidx = base + (size_t)qrow * rs + d;
      if (ALONG_W) split1(oacc[t][r], o_h, o_l, oidx);
      else         o_f[oidx] = oacc[t][r];
    }
}

// ---------------------------------------------------------------------------
// Launch
// ---------------------------------------------------------------------------
extern "C" void kernel_launch(void* const* d_in, const int* in_sizes, int n_in,
                              void* d_out, int out_size, void* d_ws,
                              size_t ws_size, hipStream_t stream) {
  const float* x      = (const float*)d_in[0];
  const float* sinp   = (const float*)d_in[1];
  const float* cosp   = (const float*)d_in[2];
  const float* mask_h = (const float*)d_in[3];
  const float* mask_w = (const float*)d_in[4];
  const float* q_w    = (const float*)d_in[5];
  const float* q_b    = (const float*)d_in[6];
  const float* k_w    = (const float*)d_in[7];
  const float* k_b    = (const float*)d_in[8];
  const float* v_w    = (const float*)d_in[9];
  const float* v_b    = (const float*)d_in[10];
  const float* lepe_w = (const float*)d_in[11];
  const float* lepe_b = (const float*)d_in[12];
  const float* out_w  = (const float*)d_in[13];
  const float* out_b  = (const float*)d_in[14];

  const size_t EB = (size_t)65536 * E_DIM;  // elements per tensor
  char* p = (char*)d_ws;
  __bf16* xh  = (__bf16*)p; p += EB * 2;    // fp32 tensors as hi/lo bf16 pairs
  __bf16* xl  = (__bf16*)p; p += EB * 2;
  __bf16* qrh = (__bf16*)p; p += EB * 2;
  __bf16* qrl = (__bf16*)p; p += EB * 2;
  __bf16* krh = (__bf16*)p; p += EB * 2;
  __bf16* krl = (__bf16*)p; p += EB * 2;
  __bf16* vh  = (__bf16*)p;                 // v region, reused as fp32 attn2
  __bf16* vl  = (__bf16*)(p + EB * 2);
  float*  attn2 = (float*)p; p += EB * 4;
  float*  lepe  = (float*)p; p += EB * 4;
  __bf16* v1h = (__bf16*)p; p += EB * 2;
  __bf16* v1l = (__bf16*)p; p += EB * 2;
  __bf16* wqh = (__bf16*)p; p += 65536 * 2;  // transposed/split weights [N][K]
  __bf16* wql = (__bf16*)p; p += 65536 * 2;
  __bf16* wkh = (__bf16*)p; p += 65536 * 2;
  __bf16* wkl = (__bf16*)p; p += 65536 * 2;
  __bf16* wvh = (__bf16*)p; p += 65536 * 2;
  __bf16* wvl = (__bf16*)p; p += 65536 * 2;
  __bf16* woh = (__bf16*)p; p += 65536 * 2;
  __bf16* wol = (__bf16*)p; p += 65536 * 2;

  const float kscale = 0.17677669529663687f;  // 32^-0.5
  dim3 gg(512, 4), gb(256);

  wprep<<<256, 256, 0, stream>>>(q_w, wqh, wql);
  wprep<<<256, 256, 0, stream>>>(k_w, wkh, wkl);
  wprep<<<256, 256, 0, stream>>>(v_w, wvh, wvl);
  wprep<<<256, 256, 0, stream>>>(out_w, woh, wol);
  split_f32<<<65536, 256, 0, stream>>>(x, nullptr, xh, xl, 0);

  gemm_bf16x3<0><<<gg, gb, 0, stream>>>(xh, xl, wqh, wql, q_b, sinp, cosp,
                                        qrh, qrl, nullptr, 1.f);
  gemm_bf16x3<1><<<gg, gb, 0, stream>>>(xh, xl, wkh, wkl, k_b, sinp, cosp,
                                        krh, krl, nullptr, kscale);
  gemm_bf16x3<2><<<gg, gb, 0, stream>>>(xh, xl, wvh, wvl, v_b, nullptr, nullptr,
                                        vh, vl, nullptr, 1.f);

  lepe_conv<<<65536, 256, 0, stream>>>(vh, vl, lepe_w, lepe_b, lepe);

  attn64<true><<<8192, 128, 0, stream>>>(qrh, qrl, krh, krl, vh, vl, mask_w,
                                         v1h, v1l, nullptr);          // along W
  attn64<false><<<8192, 128, 0, stream>>>(qrh, qrl, krh, krl, v1h, v1l, mask_h,
                                          nullptr, nullptr, attn2);   // along H

  // (attn2 + lepe) split into the now-dead x buffers, then final projection.
  split_f32<<<65536, 256, 0, stream>>>(attn2, lepe, xh, xl, 1);
  gemm_bf16x3<3><<<gg, gb, 0, stream>>>(xh, xl, woh, wol, out_b, nullptr,
                                        nullptr, nullptr, nullptr,
                                        (float*)d_out, 1.f);
}
